// BestModel5_11725260718513
// MI455X (gfx1250) — compile-verified, matrix operand order checked by reading, not run on hardware
//
#include <hip/hip_runtime.h>
#include <hip/hip_bf16.h>
#include <math.h>

// ---------------------------------------------------------------------------
// Problem constants (from reference)
// ---------------------------------------------------------------------------
#define Bsz   256
#define Tlen  40
#define Edim  256
#define Hdim  256
#define KIN   512      // E + H
#define NG    512      // 2H (gates)
#define NC    256      // H  (candidate)

typedef __attribute__((ext_vector_type(16))) __bf16 v16bf;
typedef __attribute__((ext_vector_type(8)))  float  v8f;

// ---------------------------------------------------------------------------
// Workspace layout (bytes, all 256-aligned by construction)
// ---------------------------------------------------------------------------
// Fragment-major bf16 tile store: [ktile][ntile][lane(32)][slot(16)] bf16
#define OFF_WGQ   ((size_t)0)                 // 16kt*32nt*32*16*2 = 524288
#define OFF_WGR   (OFF_WGQ + 524288)
#define OFF_WCQ   (OFF_WGR + 524288)          // 16*16*32*16*2 = 262144
#define OFF_WCR   (OFF_WCQ + 262144)
#define OFF_W1A   (OFF_WCR + 262144)          // 8*16*32*16*2 = 131072
#define OFF_W1B   (OFF_W1A + 131072)
#define OFF_XQ    (OFF_W1B + 131072)          // 16mt*40t*8kt*32*16*2 = 5242880
#define OFF_XR    (OFF_XQ  + 5242880)
#define OFF_ENCF  (OFF_XR  + 5242880)         // 2*256*256*4 = 524288
#define OFF_ENCB  (OFF_ENCF + 524288)         // 2*256*256*2 = 262144
#define OFF_PQ    (OFF_ENCB + 262144)         // 256*256*4
#define OFF_PR    (OFF_PQ  + 262144)
#define OFF_DIST  (OFF_PR  + 262144)
// total ~13.9 MB

// ---------------------------------------------------------------------------
// bf16 helpers (bit-exact RNE truncation; avoids relying on __bf16 arithmetic)
// ---------------------------------------------------------------------------
__device__ __forceinline__ __bf16 f2bf(float f) {
  unsigned u = __builtin_bit_cast(unsigned, f);
  u += 0x7FFFu + ((u >> 16) & 1u);          // round-to-nearest-even
  unsigned short h = (unsigned short)(u >> 16);
  return __builtin_bit_cast(__bf16, h);
}
__device__ __forceinline__ float bf2f(__bf16 b) {
  unsigned u = ((unsigned)__builtin_bit_cast(unsigned short, b)) << 16;
  return __builtin_bit_cast(float, u);
}

// K index held by (lane, slot) for a 16-bit 16x32 A fragment (ISA 7.12.2).
// Lanes 0-15: VGPR v<4 -> K=2v,2v+1 ; v>=4 -> K=16+2(v-4)+{0,1}. Lanes 16-31: +8.
// B fragments assumed to mirror this with column = lane&15.
__device__ __forceinline__ int kmap(int lane, int s) {
  int kh   = (lane >> 4) << 3;
  int v    = s >> 1, p = s & 1;
  int base = (v < 4) ? (2 * v) : (16 + 2 * (v - 4));
  return base + p + kh;
}

__device__ __forceinline__ v8f zero8() {
  v8f z = {0.f, 0.f, 0.f, 0.f, 0.f, 0.f, 0.f, 0.f};
  return z;
}

// ---------------------------------------------------------------------------
// Kernel 1: convert a row-major fp32 [K,N] weight to bf16 fragment-major tiles
// ---------------------------------------------------------------------------
__global__ void wfrag_kernel(const float* __restrict__ src, __bf16* __restrict__ dst,
                             int K, int N) {
  int ktiles = K >> 5, ntiles = N >> 4;
  int id = blockIdx.x * blockDim.x + threadIdx.x;
  int total = ktiles * ntiles * 32;
  if (id >= total) return;
  int lane = id & 31;
  int tile = id >> 5;
  int nt = tile % ntiles, kt = tile / ntiles;
  int n = nt * 16 + (lane & 15);
  __bf16* out = dst + (size_t)id * 16;
#pragma unroll
  for (int s = 0; s < 16; ++s) {
    int k = kt * 32 + kmap(lane, s);
    out[s] = f2bf(src[(size_t)k * N + n]);
  }
}

// ---------------------------------------------------------------------------
// Kernel 2: embedding gather -> A-fragment-major bf16 x tiles
//   dst index: [mt(16)][t(40)][ktile(8)][lane(32)][slot(16)]
// ---------------------------------------------------------------------------
__global__ void gather_kernel(const float* __restrict__ emb, const int* __restrict__ idx,
                              __bf16* __restrict__ dst) {
  int id = blockIdx.x * blockDim.x + threadIdx.x;  // 16*40*8*32 = 163840 lane-frags
  if (id >= 16 * Tlen * 8 * 32) return;
  int lane  = id & 31;
  int tile  = id >> 5;
  int ktile = tile & 7;
  int t     = (tile >> 3) % Tlen;
  int mt    = (tile >> 3) / Tlen;
  int b     = mt * 16 + (lane & 15);
  int token = idx[b * Tlen + t];
  __bf16* out = dst + (size_t)id * 16;
#pragma unroll
  for (int s = 0; s < 16; ++s) {
    int e = ktile * 32 + kmap(lane, s);
    out[s] = f2bf(emb[(size_t)token * Edim + e]);
  }
}

// ---------------------------------------------------------------------------
// Kernel 3: GRU encoder. 32 blocks: blockIdx = e*16 + mtile, 256 thr = 8 waves.
// Recurrent state lives in LDS; per step two WMMA GEMMs:
//   gates  [16x512]@[512x512] -> sigmoid -> r,z
//   cand   [16x512]@[512x256] -> tanh    -> c ;  h' = z*h + (1-z)*c (len-masked)
// Weight fragments are *streamed from L2 every step* (1.5 MB total, L2-resident
// at 192 MB). An asm-laundered zero offset keeps those loads loop-variant so
// they are neither hoisted+spilled (round-1 bug) nor demoted to flat_load by
// pointer laundering (round-2: flat ops tie up DScnt as well as LOADcnt per
// ISA 10.1.1, forcing combined waits against unrelated LDS work).
// ---------------------------------------------------------------------------
__global__ __launch_bounds__(256) void gru_kernel(
    const __bf16* __restrict__ xq, const __bf16* __restrict__ xr,
    const __bf16* __restrict__ wgq, const __bf16* __restrict__ wgr,
    const __bf16* __restrict__ wcq, const __bf16* __restrict__ wcr,
    const float* __restrict__ bgq, const float* __restrict__ bcq,
    const float* __restrict__ bgr, const float* __restrict__ bcr,
    const int* __restrict__ qlen, const int* __restrict__ rlen,
    float* __restrict__ encf, __bf16* __restrict__ encb) {
  __shared__ float  hsf[16][260];   // fp32 master hidden state (padded)
  __shared__ float  zs [16][260];   // update gate z (fp32)
  __shared__ __bf16 hb [16][256];   // bf16 operand copy of h
  __shared__ __bf16 rhb[16][256];   // bf16 r*h operand
  __shared__ float  bgl[512];       // gate bias (LDS-resident)
  __shared__ float  bcl[256];       // candidate bias
  __shared__ int    len_s[16];

  const int e  = blockIdx.x >> 4;
  const int mt = blockIdx.x & 15;
  const int wid = threadIdx.x >> 5, lane = threadIdx.x & 31;
  const int rbase = (lane >> 4) << 3;   // C/D row base per lane half
  const int nlo   = lane & 15;

  const __bf16* xfrag = e ? xr : xq;
  const __bf16* wgb   = e ? wgr : wgq;
  const __bf16* wcb   = e ? wcr : wcq;
  const float*  bg    = e ? bgr : bgq;
  const float*  bc    = e ? bcr : bcq;
  const int*    lens  = e ? rlen : qlen;

  for (int i = threadIdx.x; i < 16 * 256; i += 256) {
    int rm = i >> 8, n = i & 255;
    hsf[rm][n] = 0.f;
    hb[rm][n]  = f2bf(0.f);
  }
  {
    int tid = threadIdx.x;
    bgl[tid]        = bg[tid];
    bgl[tid + 256]  = bg[tid + 256];
    bcl[tid]        = bc[tid];
    if (tid < 16) len_s[tid] = lens[mt * 16 + tid];
  }
  __syncthreads();

  // Always-zero element offset the compiler must treat as changing every step:
  // keeps the 96 weight-fragment loads per step inside the t-loop while the
  // pointers retain global-address-space provenance (=> global_load_b128).
  size_t zoff = 0;

#pragma unroll 1
  for (int t = 0; t < Tlen; ++t) {
    asm volatile("" : "+s"(zoff));
    const __bf16* wg = wgb + zoff;
    const __bf16* wc = wcb + zoff;

    // ---- gates GEMM: each wave owns 4 N-tiles (64 cols) of the 512 outputs
    v8f gacc[4];
#pragma unroll
    for (int tt = 0; tt < 4; ++tt) gacc[tt] = zero8();

    for (int kt = 0; kt < 16; ++kt) {
      v16bf a;
      if (kt < 8) {
        const __bf16* xp = xfrag + ((((size_t)mt * Tlen + t) * 8 + kt) * 32 + lane) * 16;
        a = *(const v16bf*)xp;
      } else {
        int row = lane & 15, koff = (kt - 8) * 32;
#pragma unroll
        for (int s = 0; s < 16; ++s) a[s] = hb[row][koff + kmap(lane, s)];
      }
#pragma unroll
      for (int tt = 0; tt < 4; ++tt) {
        int nt = wid * 4 + tt;
        v16bf b = *(const v16bf*)(wg + (((size_t)kt * 32 + nt) * 32 + lane) * 16);
        gacc[tt] = __builtin_amdgcn_wmma_f32_16x16x32_bf16(
            false, a, false, b, (short)0, gacc[tt], false, false);
      }
    }

    // ---- sigmoid; waves 0-3 produce r*h operand, waves 4-7 stash z
#pragma unroll
    for (int tt = 0; tt < 4; ++tt) {
      int col = (wid * 4 + tt) * 16 + nlo;       // global gate column 0..511
#pragma unroll
      for (int i = 0; i < 8; ++i) {
        int row = i + rbase;
        float g  = gacc[tt][i] + bgl[col];
        float sg = 1.f / (1.f + expf(-g));
        if (col < 256) rhb[row][col] = f2bf(sg * hsf[row][col]);
        else           zs[row][col - 256] = sg;
      }
    }
    __syncthreads();

    // ---- candidate GEMM: each wave owns 2 N-tiles (32 cols) of 256 outputs
    v8f cacc[2];
    cacc[0] = zero8(); cacc[1] = zero8();
    for (int kt = 0; kt < 16; ++kt) {
      v16bf a;
      if (kt < 8) {
        const __bf16* xp = xfrag + ((((size_t)mt * Tlen + t) * 8 + kt) * 32 + lane) * 16;
        a = *(const v16bf*)xp;
      } else {
        int row = lane & 15, koff = (kt - 8) * 32;
#pragma unroll
        for (int s = 0; s < 16; ++s) a[s] = rhb[row][koff + kmap(lane, s)];
      }
#pragma unroll
      for (int tt = 0; tt < 2; ++tt) {
        int nt = wid * 2 + tt;
        v16bf b = *(const v16bf*)(wc + (((size_t)kt * 16 + nt) * 32 + lane) * 16);
        cacc[tt] = __builtin_amdgcn_wmma_f32_16x16x32_bf16(
            false, a, false, b, (short)0, cacc[tt], false, false);
      }
    }

    // ---- state update (each (row,col) touched by exactly one lane)
#pragma unroll
    for (int tt = 0; tt < 2; ++tt) {
      int col = (wid * 2 + tt) * 16 + nlo;
#pragma unroll
      for (int i = 0; i < 8; ++i) {
        int row = i + rbase;
        float c  = tanhf(cacc[tt][i] + bcl[col]);
        float ho = hsf[row][col];
        float z  = zs[row][col];
        float hn = z * ho + (1.f - z) * c;
        float hf = (t < len_s[row]) ? hn : ho;
        hsf[row][col] = hf;
        hb[row][col]  = f2bf(hf);
      }
    }
    __syncthreads();
  }

  for (int i = threadIdx.x; i < 16 * 256; i += 256) {
    int rm = i >> 8, n = i & 255;
    size_t o = ((size_t)e * 256 + mt * 16 + rm) * 256 + n;
    encf[o] = hsf[rm][n];
    encb[o] = f2bf(hsf[rm][n]);
  }
}

// ---------------------------------------------------------------------------
// Kernel 4: Pq = Qenc@W1a, Pr = Renc@W1b, dist = Qenc@Renc^T (all WMMA bf16)
// 96 blocks x 8 waves = 768 output tiles (3 jobs x 16x16 tiles)
// ---------------------------------------------------------------------------
__global__ __launch_bounds__(256) void proj_kernel(
    const __bf16* __restrict__ encb, const __bf16* __restrict__ w1a,
    const __bf16* __restrict__ w1b, float* __restrict__ pq,
    float* __restrict__ pr, float* __restrict__ dist) {
  int wid = threadIdx.x >> 5, lane = threadIdx.x & 31;
  int tile = blockIdx.x * 8 + wid;
  int job = tile >> 8, rem = tile & 255;
  int mt = rem >> 4, nt = rem & 15;
  const __bf16* qenc = encb;
  const __bf16* renc = encb + 256 * 256;
  const __bf16* aenc = (job == 1) ? renc : qenc;

  v8f acc = zero8();
  int mrow = mt * 16 + (lane & 15);
  for (int kt = 0; kt < 8; ++kt) {
    v16bf a, b;
#pragma unroll
    for (int s = 0; s < 16; ++s)
      a[s] = aenc[(size_t)mrow * 256 + kt * 32 + kmap(lane, s)];
    if (job == 2) {
      int nrow = nt * 16 + (lane & 15);
#pragma unroll
      for (int s = 0; s < 16; ++s)
        b[s] = renc[(size_t)nrow * 256 + kt * 32 + kmap(lane, s)];
    } else {
      const __bf16* wp = (job == 0 ? w1a : w1b) + (((size_t)kt * 16 + nt) * 32 + lane) * 16;
      b = *(const v16bf*)wp;
    }
    acc = __builtin_amdgcn_wmma_f32_16x16x32_bf16(
        false, a, false, b, (short)0, acc, false, false);
  }
  float* out = (job == 0) ? pq : (job == 1 ? pr : dist);
  int n = nt * 16 + (lane & 15);
#pragma unroll
  for (int i = 0; i < 8; ++i) {
    int row = mt * 16 + i + ((lane >> 4) << 3);
    out[(size_t)row * 256 + n] = acc[i];
  }
}

// ---------------------------------------------------------------------------
// Kernel 5: head. h = gelu(Pq[qi] + Pr[ri] + d*w1d + b1); logits = h@W2 + b2
// One block per qi; 8 waves sweep ri; wave-shuffle reduction (wave32).
// ---------------------------------------------------------------------------
__global__ __launch_bounds__(256) void head_kernel(
    const float* __restrict__ pq, const float* __restrict__ pr,
    const float* __restrict__ dist, const float* __restrict__ W1,
    const float* __restrict__ b1, const float* __restrict__ W2,
    const float* __restrict__ b2, float* __restrict__ out) {
  __shared__ float pqs[256], drow[256], w1d[256], b1s[256], w2a[256], w2b[256];
  int qi = blockIdx.x, tid = threadIdx.x;
  pqs[tid]  = pq[(size_t)qi * 256 + tid];
  drow[tid] = dist[(size_t)qi * 256 + tid];
  w1d[tid]  = W1[(size_t)256 * 256 + tid];   // row 256 of [513,256] W1
  b1s[tid]  = b1[tid];
  w2a[tid]  = W2[tid * 2 + 0];
  w2b[tid]  = W2[tid * 2 + 1];
  __syncthreads();
  float bb0 = b2[0], bb1 = b2[1];
  int wid = tid >> 5, lane = tid & 31;

  for (int ri = wid; ri < 256; ri += 8) {
    float d = drow[ri];
    const float* prow = pr + (size_t)ri * 256;
    float s0 = 0.f, s1 = 0.f;
#pragma unroll
    for (int c = 0; c < 8; ++c) {
      int n = lane + 32 * c;
      float x = pqs[n] + prow[n] + d * w1d[n] + b1s[n];
      float g = 0.5f * x * (1.f + tanhf(0.7978845608028654f * (x + 0.044715f * x * x * x)));
      s0 += g * w2a[n];
      s1 += g * w2b[n];
    }
#pragma unroll
    for (int off = 16; off >= 1; off >>= 1) {
      s0 += __shfl_down(s0, off, 32);
      s1 += __shfl_down(s1, off, 32);
    }
    if (lane == 0) {
      int row = (ri == qi) ? qi : (256 + qi * 255 + (ri < qi ? ri : ri - 1));
      out[(size_t)row * 2 + 0] = s0 + bb0;
      out[(size_t)row * 2 + 1] = s1 + bb1;
    }
  }
}

// ---------------------------------------------------------------------------
extern "C" void kernel_launch(void* const* d_in, const int* in_sizes, int n_in,
                              void* d_out, int out_size, void* d_ws, size_t ws_size,
                              hipStream_t stream) {
  const int*   iq   = (const int*)d_in[0];
  const int*   ir   = (const int*)d_in[1];
  const int*   qlen = (const int*)d_in[2];
  const int*   rlen = (const int*)d_in[3];
  const float* emb  = (const float*)d_in[4];
  const float* Wg_q = (const float*)d_in[5];
  const float* bg_q = (const float*)d_in[6];
  const float* Wc_q = (const float*)d_in[7];
  const float* bc_q = (const float*)d_in[8];
  const float* Wg_r = (const float*)d_in[9];
  const float* bg_r = (const float*)d_in[10];
  const float* Wc_r = (const float*)d_in[11];
  const float* bc_r = (const float*)d_in[12];
  const float* W1   = (const float*)d_in[13];
  const float* b1   = (const float*)d_in[14];
  const float* W2   = (const float*)d_in[15];
  const float* b2   = (const float*)d_in[16];
  float* out = (float*)d_out;

  char* ws = (char*)d_ws;
  __bf16* wgq  = (__bf16*)(ws + OFF_WGQ);
  __bf16* wgr  = (__bf16*)(ws + OFF_WGR);
  __bf16* wcq  = (__bf16*)(ws + OFF_WCQ);
  __bf16* wcr  = (__bf16*)(ws + OFF_WCR);
  __bf16* w1a  = (__bf16*)(ws + OFF_W1A);
  __bf16* w1b  = (__bf16*)(ws + OFF_W1B);
  __bf16* xq   = (__bf16*)(ws + OFF_XQ);
  __bf16* xr   = (__bf16*)(ws + OFF_XR);
  float*  encf = (float*)(ws + OFF_ENCF);
  __bf16* encb = (__bf16*)(ws + OFF_ENCB);
  float*  pq   = (float*)(ws + OFF_PQ);
  float*  pr   = (float*)(ws + OFF_PR);
  float*  dist = (float*)(ws + OFF_DIST);

  // 1) weights -> bf16 fragment tiles
  wfrag_kernel<<<(16 * 32 * 32 + 255) / 256, 256, 0, stream>>>(Wg_q, wgq, KIN, NG);
  wfrag_kernel<<<(16 * 32 * 32 + 255) / 256, 256, 0, stream>>>(Wg_r, wgr, KIN, NG);
  wfrag_kernel<<<(16 * 16 * 32 + 255) / 256, 256, 0, stream>>>(Wc_q, wcq, KIN, NC);
  wfrag_kernel<<<(16 * 16 * 32 + 255) / 256, 256, 0, stream>>>(Wc_r, wcr, KIN, NC);
  wfrag_kernel<<<(8 * 16 * 32 + 255) / 256, 256, 0, stream>>>(W1, w1a, 256, 256);
  wfrag_kernel<<<(8 * 16 * 32 + 255) / 256, 256, 0, stream>>>(W1 + (size_t)257 * 256, w1b, 256, 256);

  // 2) embedding gather -> A-fragment x tiles
  gather_kernel<<<(16 * Tlen * 8 * 32 + 255) / 256, 256, 0, stream>>>(emb, iq, xq);
  gather_kernel<<<(16 * Tlen * 8 * 32 + 255) / 256, 256, 0, stream>>>(emb, ir, xr);

  // 3) GRU encoders (both in one grid: 2 encoders x 16 batch tiles)
  gru_kernel<<<32, 256, 0, stream>>>(xq, xr, wgq, wgr, wcq, wcr,
                                     bg_q, bc_q, bg_r, bc_r, qlen, rlen, encf, encb);

  // 4) Pq / Pr / dist via WMMA
  proj_kernel<<<96, 256, 0, stream>>>(encb, w1a, w1b, pq, pr, dist);

  // 5) pairwise head
  head_kernel<<<256, 256, 0, stream>>>(pq, pr, dist, W1, b1, W2, b2, out);
}
